// CustonDATransformer_59356448031002
// MI455X (gfx1250) — matrix operand order repaired and checked by reading, not compiled, not run on hardware
//
#include <hip/hip_runtime.h>
#include <math.h>

// ---------------- problem constants ----------------
#define BB      2
#define CC      512
#define HH      128
#define WW      128
#define NN      (HH*WW)          // 16384
#define N_LOG   14
#define C_LOG   9
#define L_NUM   6
#define HEADS   8
#define POINTS  4
#define HEAD_D  64
#define M_ROWS  (BB*NN)          // 32768

typedef unsigned short ushortx;
typedef __attribute__((ext_vector_type(16))) __bf16 v16bf;
typedef __attribute__((ext_vector_type(8)))  float  v8f;

union FragBF { unsigned u[8]; v16bf v; };

// gfx1250 async global->LDS path (ASYNCcnt-tracked), with safe fallback
#if defined(__gfx1250__) && __has_builtin(__builtin_amdgcn_global_load_async_to_lds_b128)
#define USE_ASYNC_LDS 1
#else
#define USE_ASYNC_LDS 0
#endif

#if USE_ASYNC_LDS
typedef int v4i_ __attribute__((vector_size(16)));
typedef __attribute__((address_space(1))) v4i_* gvec_t;   // global (AS1)
typedef __attribute__((address_space(3))) v4i_* lvec_t;   // LDS (AS3)
#endif

static __device__ __forceinline__ void async_cp16(const ushortx* g, ushortx* l) {
#if USE_ASYNC_LDS
    v4i_* gg = (v4i_*)const_cast<ushortx*>(g);    // retype in generic AS
    v4i_* ll = (v4i_*)l;
    __builtin_amdgcn_global_load_async_to_lds_b128((gvec_t)gg, (lvec_t)ll, 0, 0);
#else
    (void)g; (void)l;
#endif
}
static __device__ __forceinline__ void wait_async0() {
#if __has_builtin(__builtin_amdgcn_s_wait_asynccnt)
    __builtin_amdgcn_s_wait_asynccnt(0);
#else
    asm volatile("s_wait_asynccnt 0x0" ::: "memory");
#endif
}

static __device__ __forceinline__ ushortx f2bf(float f) {
    union { float f; unsigned u; } c; c.f = f;
    unsigned u = c.u;
    unsigned r = u + 0x7fffu + ((u >> 16) & 1u);   // round-to-nearest-even
    return (ushortx)(r >> 16);
}

// =====================================================================
// bf16 WMMA GEMM, bf16 A[M,K] row-major, bf16 Bt[Nc,K] (pre-transposed).
// block tile 128x128x32, 8 waves (4x2), wave tile 32x64 = 2x4 WMMA tiles.
// Double-buffered LDS fed by async global->LDS b128 copies (fallback: regs).
// =====================================================================
template<bool RELU, bool HASB, bool HASR, bool OUTBF>
__global__ __launch_bounds__(256) void gemm_bf16_kernel(
    const ushortx* __restrict__ A, const ushortx* __restrict__ Bt,
    const float* __restrict__ bias, const float* __restrict__ res,
    float* __restrict__ DstF, ushortx* __restrict__ DstH,
    int M, int K, int Nc)
{
    __shared__ ushortx sA[2][128][40];   // [buf][m][k], padded rows
    __shared__ ushortx sB[2][128][40];   // [buf][n][k]

    const int tid  = threadIdx.x;
    const int wid  = tid >> 5, lane = tid & 31;
    const int wm   = wid & 3,  wn   = wid >> 2;     // 4x2 wave grid
    const int lrow = lane & 15, hlf = lane >> 4;

    const int m0 = blockIdx.x * 128, n0 = blockIdx.y * 128;

    // staging: identical pattern for A and B: 128 rows x 2 halves of 16
    const int srow = tid >> 1, sseg = (tid & 1) << 4;
    const ushortx* aBase = A  + (size_t)(m0 + srow) * K + sseg;
    const ushortx* bBase = Bt + (size_t)(n0 + srow) * K + sseg;

#if USE_ASYNC_LDS
    auto issue = [&](int buf, int k0) {
        async_cp16(aBase + k0,     &sA[buf][srow][sseg]);
        async_cp16(aBase + k0 + 8, &sA[buf][srow][sseg + 8]);
        async_cp16(bBase + k0,     &sB[buf][srow][sseg]);
        async_cp16(bBase + k0 + 8, &sB[buf][srow][sseg + 8]);
    };
#else
    uint4 rA0, rA1, rB0, rB1;
    auto gload = [&](int k0) {
        rA0 = *(const uint4*)(aBase + k0);
        rA1 = *(const uint4*)(aBase + k0 + 8);
        rB0 = *(const uint4*)(bBase + k0);
        rB1 = *(const uint4*)(bBase + k0 + 8);
        if (k0 + 32 < K) __builtin_prefetch(aBase + k0 + 32, 0, 1);
    };
    auto sstore = [&](int buf) {
        *(uint4*)&sA[buf][srow][sseg]     = rA0;
        *(uint4*)&sA[buf][srow][sseg + 8] = rA1;
        *(uint4*)&sB[buf][srow][sseg]     = rB0;
        *(uint4*)&sB[buf][srow][sseg + 8] = rB1;
    };
#endif

    v8f acc[2][4] = {};

#if USE_ASYNC_LDS
    issue(0, 0); wait_async0();
#else
    gload(0); sstore(0);
#endif
    __syncthreads();

    const int NIT = K >> 5;
    for (int it = 0; it < NIT; ++it) {
        const int cur = it & 1;
        const bool hasNext = (it + 1) < NIT;
#if USE_ASYNC_LDS
        if (hasNext) issue(cur ^ 1, (it + 1) << 5);   // async copies in flight
#else
        if (hasNext) gload((it + 1) << 5);
#endif

        // ---- fragments per ISA 7.12.2 lane layouts ----
        FragBF a[2], b[4];
        #pragma unroll
        for (int ti = 0; ti < 2; ++ti) {
            const int m = wm * 32 + ti * 16 + lrow;
            #pragma unroll
            for (int vv = 0; vv < 8; ++vv) {
                const int kk = ((vv >> 2) << 4) + hlf * 8 + ((vv & 3) << 1);
                a[ti].u[vv] = *(const unsigned*)&sA[cur][m][kk];
            }
        }
        #pragma unroll
        for (int tj = 0; tj < 4; ++tj) {
            const int n = wn * 64 + tj * 16 + lrow;
            #pragma unroll
            for (int vv = 0; vv < 8; ++vv) {
                const int kk = ((vv >> 2) << 4) + hlf * 8 + ((vv & 3) << 1);
                b[tj].u[vv] = *(const unsigned*)&sB[cur][n][kk];
            }
        }
        #pragma unroll
        for (int ti = 0; ti < 2; ++ti)
            #pragma unroll
            for (int tj = 0; tj < 4; ++tj)
                acc[ti][tj] = __builtin_amdgcn_wmma_f32_16x16x32_bf16(
                    false, a[ti].v, false, b[tj].v, (short)0, acc[ti][tj],
                    false, false);

#if USE_ASYNC_LDS
        if (hasNext) wait_async0();                   // ASYNCcnt -> 0
#else
        if (hasNext) sstore(cur ^ 1);
#endif
        __syncthreads();
    }

    // ---- epilogue (C/D layout: lane%16 = N, vgpr r + 8*(lane/16) = M) ----
    #pragma unroll
    for (int ti = 0; ti < 2; ++ti) {
        #pragma unroll
        for (int tj = 0; tj < 4; ++tj) {
            const int rbase = m0 + wm * 32 + ti * 16 + hlf * 8;
            const int col   = n0 + wn * 64 + tj * 16 + lrow;
            #pragma unroll
            for (int r = 0; r < 8; ++r) {
                const int row = rbase + r;
                float val = acc[ti][tj][r];
                if (HASB) val += bias[col];
                if (HASR) val += res[(size_t)row * Nc + col];
                if (RELU) val = fmaxf(val, 0.0f);
                if (OUTBF) DstH[(size_t)row * Nc + col] = f2bf(val);
                else       DstF[(size_t)row * Nc + col] = val;
            }
        }
    }
}

// =====================================================================
// weight prep: f32 (L, K=512, 512) -> bf16 transposed (L, 512, 512)
// =====================================================================
__global__ void wtrans_kernel(const float* __restrict__ W, ushortx* __restrict__ Wt) {
    __shared__ float tile[32][33];
    const int z = blockIdx.z;
    const int r0 = blockIdx.x * 32, c0 = blockIdx.y * 32;   // r = k, c = n
    const int tx = threadIdx.x, ty = threadIdx.y;
    const float* Wi = W + (size_t)z * CC * CC;
    ushortx* Wo = Wt + (size_t)z * CC * CC;
    #pragma unroll
    for (int k = 0; k < 32; k += 8)
        tile[ty + k][tx] = Wi[(size_t)(r0 + ty + k) * CC + c0 + tx];
    __syncthreads();
    #pragma unroll
    for (int k = 0; k < 32; k += 8)
        Wo[(size_t)(c0 + ty + k) * CC + r0 + tx] = f2bf(tile[tx][ty + k]);
}

// concat Woff(C,64)|Waw(C,32)|pad -> bf16 transposed Wcatt(L,128,512)
__global__ void wcat_kernel(const float* __restrict__ Woff, const float* __restrict__ Waw,
                            ushortx* __restrict__ Wcatt) {
    const int idx = blockIdx.x * 256 + threadIdx.x;   // over L*128*512
    const int l = idx / (128 * CC), rem = idx % (128 * CC);
    const int col = rem >> 9, k = rem & (CC - 1);
    float val = 0.0f;
    if (col < 64)      val = Woff[(size_t)l * CC * 64 + (size_t)k * 64 + col];
    else if (col < 96) val = Waw [(size_t)l * CC * 32 + (size_t)k * 32 + (col - 64)];
    Wcatt[idx] = f2bf(val);
}
__global__ void bcat_kernel(const float* __restrict__ boff, const float* __restrict__ baw,
                            float* __restrict__ bcat) {
    const int idx = blockIdx.x * 256 + threadIdx.x;   // over L*128
    if (idx >= L_NUM * 128) return;
    const int l = idx >> 7, col = idx & 127;
    float val = 0.0f;
    if (col < 64)      val = boff[l * 64 + col];
    else if (col < 96) val = baw [l * 32 + (col - 64)];
    bcat[idx] = val;
}

// =====================================================================
// voxelization: mask, stable valid-first ordering, packing
// =====================================================================
__global__ void mask_kernel(const float* __restrict__ lidar, int* __restrict__ mask) {
    const int idx = blockIdx.x * 256 + threadIdx.x;   // over B*N
    const int b = idx >> N_LOG, n = idx & (NN - 1);
    const size_t base = (size_t)b * CC * NN + n;
    int ok = 1;
    for (int c = 0; c < CC; ++c) {
        if (lidar[base + (size_t)c * NN] == 0.0f) { ok = 0; break; }
    }
    mask[idx] = ok;
}

__global__ __launch_bounds__(1024) void order_kernel(
    const int* __restrict__ mask, int* __restrict__ order, int* __restrict__ numValid)
{
    const int b = blockIdx.x, tid = threadIdx.x;
    const int PER = NN / 1024;                    // 16
    __shared__ int cnt[1024];
    __shared__ int totalS;
    int c = 0;
    #pragma unroll
    for (int j = 0; j < PER; ++j) c += mask[b * NN + tid * PER + j];
    cnt[tid] = c;
    __syncthreads();
    for (int off = 1; off < 1024; off <<= 1) {    // inclusive Hillis-Steele scan
        int t = (tid >= off) ? cnt[tid - off] : 0;
        __syncthreads();
        cnt[tid] += t;
        __syncthreads();
    }
    if (tid == 1023) totalS = cnt[1023];
    __syncthreads();
    int running = cnt[tid] - c;                   // exclusive prefix of valid
    const int nv = totalS;
    for (int j = 0; j < PER; ++j) {
        const int n = tid * PER + j;
        if (mask[b * NN + n]) { order[b * NN + running] = n; ++running; }
        else                  { order[b * NN + nv + (n - running)] = n; }
    }
    if (tid == 0) numValid[b] = nv;
}

__global__ void pack_q_kernel(const float* __restrict__ lidar,
                              const int* __restrict__ order,
                              const int* __restrict__ numValid,
                              float* __restrict__ q)
{
    const int idx = blockIdx.x * 256 + threadIdx.x;  // over B*N*C
    const int c = idx & (CC - 1), bn = idx >> C_LOG;
    const int b = bn >> N_LOG, pos = bn & (NN - 1);
    const int n = order[bn];
    const float v = lidar[(size_t)b * CC * NN + (size_t)c * NN + n];
    q[idx] = (pos < numValid[b]) ? v : 0.0f;
}

__global__ void ref_kernel(const int* __restrict__ order,
                           const int* __restrict__ numValid,
                           float* __restrict__ ref)
{
    const int bn = blockIdx.x * 256 + threadIdx.x;   // over B*N
    const int b = bn >> N_LOG, pos = bn & (NN - 1);
    const int n = order[bn];
    const float val = (pos < numValid[b]) ? 1.0f : 0.0f;
    ref[(size_t)bn * 2 + 0] = val * (float)(n & (WW - 1)) * (1.0f / WW);
    ref[(size_t)bn * 2 + 1] = val * (float)(n >> 7)       * (1.0f / HH);
}

// bev (B,C,N) f32 -> feat_flat (B,N,C) bf16, LDS tiled transpose
__global__ void bevtobf_kernel(const float* __restrict__ bev, ushortx* __restrict__ ff) {
    __shared__ float tile[32][33];
    const int b = blockIdx.z;
    const int n0 = blockIdx.x * 32, c0 = blockIdx.y * 32;
    const int tx = threadIdx.x, ty = threadIdx.y;
    #pragma unroll
    for (int k = 0; k < 32; k += 8)
        tile[ty + k][tx] = bev[(size_t)b * CC * NN + (size_t)(c0 + ty + k) * NN + n0 + tx];
    __syncthreads();
    #pragma unroll
    for (int k = 0; k < 32; k += 8)
        ff[(size_t)b * NN * CC + (size_t)(n0 + ty + k) * CC + c0 + tx] = f2bf(tile[tx][ty + k]);
}

// positional embedding: relu(BN(ref @ pe_W1)) -> bf16 (B,N,C)
__global__ void pe_kernel(const float* __restrict__ ref, const float* __restrict__ peW1,
                          const float* __restrict__ g, const float* __restrict__ bt,
                          const float* __restrict__ m, const float* __restrict__ v,
                          ushortx* __restrict__ dst)
{
    const int idx = blockIdx.x * 256 + threadIdx.x;  // over B*N*C
    const int c = idx & (CC - 1), bn = idx >> C_LOG;
    const float rx = ref[(size_t)bn * 2], ry = ref[(size_t)bn * 2 + 1];
    float p = rx * peW1[c] + ry * peW1[CC + c];
    p = (p - m[c]) * rsqrtf(v[c] + 1e-5f) * g[c] + bt[c];
    dst[idx] = f2bf(fmaxf(p, 0.0f));
}

// x = q + q_pos, emitted as bf16 (only consumed as GEMM A)
__global__ void add_kernel(const float* __restrict__ a, const float* __restrict__ b,
                           ushortx* __restrict__ d) {
    const int idx = blockIdx.x * 256 + threadIdx.x;
    d[idx] = f2bf(a[idx] + b[idx]);
}

// =====================================================================
// fused softmax + bilinear deformable sampling; bf16 output
// one block per token; thread t -> head t/32, dim pair (t%32)*2
// =====================================================================
__global__ __launch_bounds__(256) void sample_kernel(
    const float* __restrict__ v, const float* __restrict__ offaw,
    const float* __restrict__ ref, ushortx* __restrict__ out)
{
    const int bn = blockIdx.x;
    const int b = bn >> N_LOG;
    const int t = threadIdx.x;
    const int head = t >> 5, d0 = (t & 31) << 1;
    const float* row = offaw + (size_t)bn * 128;

    float lg[POINTS], e[POINTS];
    float mx = -1e30f;
    #pragma unroll
    for (int p = 0; p < POINTS; ++p) { lg[p] = row[64 + head * 4 + p]; mx = fmaxf(mx, lg[p]); }
    float s = 0.0f;
    #pragma unroll
    for (int p = 0; p < POINTS; ++p) { e[p] = __expf(lg[p] - mx); s += e[p]; }
    const float inv = 1.0f / s;

    const float rx = ref[(size_t)bn * 2], ry = ref[(size_t)bn * 2 + 1];
    const float* vb = v + ((size_t)b << N_LOG) * CC + head * HEAD_D + d0;

    float ax = 0.0f, ay = 0.0f;
    #pragma unroll
    for (int p = 0; p < POINTS; ++p) {
        const float ox = row[head * 8 + p * 2], oy = row[head * 8 + p * 2 + 1];
        const float fx = rx * (float)WW + ox - 0.5f;   // (rx + ox/W)*W - 0.5
        const float fy = ry * (float)HH + oy - 0.5f;
        const float x0f = floorf(fx), y0f = floorf(fy);
        const float wx = fx - x0f, wy = fy - y0f;
        const int x0 = (int)x0f, y0 = (int)y0f;
        const float awp = e[p] * inv;
        const int   cx[4] = { x0, x0 + 1, x0,     x0 + 1 };
        const int   cy[4] = { y0, y0,     y0 + 1, y0 + 1 };
        const float cw[4] = { (1 - wx) * (1 - wy), wx * (1 - wy),
                              (1 - wx) * wy,       wx * wy };
        #pragma unroll
        for (int k = 0; k < 4; ++k) {
            const int xi = cx[k], yi = cy[k];
            if (xi >= 0 && xi < WW && yi >= 0 && yi < HH) {
                const float2 g = *(const float2*)(vb + (size_t)(yi * WW + xi) * CC);
                const float w = awp * cw[k];
                ax += w * g.x; ay += w * g.y;
            }
        }
    }
    ushortx* o = out + (size_t)bn * CC + head * HEAD_D + d0;
    o[0] = f2bf(ax); o[1] = f2bf(ay);
}

// =====================================================================
// layernorm over C=512 per row; writes f32 + bf16 shadow
// =====================================================================
__global__ __launch_bounds__(256) void layernorm_kernel(
    const float* __restrict__ src, const float* __restrict__ g,
    const float* __restrict__ bt, float* __restrict__ dst,
    ushortx* __restrict__ dsth)
{
    const int row = blockIdx.x, tid = threadIdx.x;
    const float* s = src + (size_t)row * CC;
    const float a0 = s[tid], a1 = s[tid + 256];
    __shared__ float r1[256], r2[256];
    r1[tid] = a0 + a1; r2[tid] = a0 * a0 + a1 * a1;
    __syncthreads();
    for (int off = 128; off > 0; off >>= 1) {
        if (tid < off) { r1[tid] += r1[tid + off]; r2[tid] += r2[tid + off]; }
        __syncthreads();
    }
    const float mean = r1[0] * (1.0f / CC);
    const float var  = r2[0] * (1.0f / CC) - mean * mean;
    const float rstd = rsqrtf(var + 1e-5f);
    const float o0 = (a0 - mean) * rstd * g[tid]       + bt[tid];
    const float o1 = (a1 - mean) * rstd * g[tid + 256] + bt[tid + 256];
    float* d = dst + (size_t)row * CC;
    d[tid] = o0; d[tid + 256] = o1;
    ushortx* dh = dsth + (size_t)row * CC;
    dh[tid] = f2bf(o0); dh[tid + 256] = f2bf(o1);
}

// scatter packed q back to dense (B,C,H,W); permutation covers all n
__global__ void scatter_kernel(const float* __restrict__ q, const int* __restrict__ order,
                               const int* __restrict__ numValid, float* __restrict__ out)
{
    const int idx = blockIdx.x * 256 + threadIdx.x;  // over B*N*C
    const int c = idx & (CC - 1), bn = idx >> C_LOG;
    const int b = bn >> N_LOG, pos = bn & (NN - 1);
    const int n = order[bn];
    const float val = (pos < numValid[b]) ? q[idx] : 0.0f;
    out[(size_t)b * CC * NN + (size_t)c * NN + n] = val;
}

// =====================================================================
extern "C" void kernel_launch(void* const* d_in, const int* in_sizes, int n_in,
                              void* d_out, int out_size, void* d_ws, size_t ws_size,
                              hipStream_t stream)
{
    const float* bev   = (const float*)d_in[0];
    const float* lidar = (const float*)d_in[1];
    const float* Wv    = (const float*)d_in[2];
    const float* bv    = (const float*)d_in[3];
    const float* Woff  = (const float*)d_in[4];
    const float* boff  = (const float*)d_in[5];
    const float* Waw   = (const float*)d_in[6];
    const float* baw   = (const float*)d_in[7];
    const float* Wout  = (const float*)d_in[8];
    const float* bout  = (const float*)d_in[9];
    const float* ln1g  = (const float*)d_in[10];
    const float* ln1b  = (const float*)d_in[11];
    const float* W1    = (const float*)d_in[12];
    const float* W2    = (const float*)d_in[13];
    const float* ln2g  = (const float*)d_in[14];
    const float* ln2b  = (const float*)d_in[15];
    const float* peW1  = (const float*)d_in[16];
    const float* bng   = (const float*)d_in[17];
    const float* bnb   = (const float*)d_in[18];
    const float* bnm   = (const float*)d_in[19];
    const float* bnv   = (const float*)d_in[20];
    const float* peW2  = (const float*)d_in[21];
    float* out = (float*)d_out;

    // ---- workspace carve ----
    char* p = (char*)d_ws;
    auto carve = [&](size_t bytes) -> void* {
        void* r = (void*)p;
        p += (bytes + 255) & ~(size_t)255;
        return r;
    };
    const size_t SZ_BNC = (size_t)M_ROWS * CC;          // 16.8M elements
    const size_t SZ_W   = (size_t)L_NUM * CC * CC;
    float*   qbuf   = (float*)  carve(SZ_BNC * 4);
    float*   qpos   = (float*)  carve(SZ_BNC * 4);
    float*   vbuf   = (float*)  carve(SZ_BNC * 4);      // v + pre-LN scratch
    ushortx* featbf = (ushortx*)carve(SZ_BNC * 2);
    ushortx* qh     = (ushortx*)carve(SZ_BNC * 2);      // bf16 shadow of q
    ushortx* xh     = (ushortx*)carve(SZ_BNC * 2);      // pe-relu / x / FFN hidden
    ushortx* samph  = (ushortx*)carve(SZ_BNC * 2);
    float*   offaw  = (float*)  carve((size_t)M_ROWS * 128 * 4);
    ushortx* Wvt    = (ushortx*)carve(SZ_W * 2);
    ushortx* Woutt  = (ushortx*)carve(SZ_W * 2);
    ushortx* W1t    = (ushortx*)carve(SZ_W * 2);
    ushortx* W2t    = (ushortx*)carve(SZ_W * 2);
    ushortx* peW2t  = (ushortx*)carve((size_t)CC * CC * 2);
    ushortx* Wcatt  = (ushortx*)carve((size_t)L_NUM * 128 * CC * 2);
    float*   bcat   = (float*)  carve((size_t)L_NUM * 128 * 4);
    float*   refb   = (float*)  carve((size_t)M_ROWS * 2 * 4);
    int*     order  = (int*)    carve((size_t)M_ROWS * 4);
    int*     maskb  = (int*)    carve((size_t)M_ROWS * 4);
    int*     numV   = (int*)    carve(64 * 4);

    const int EW = (M_ROWS * CC) / 256;                 // elementwise grid (65536)
    const int BN_BLK = M_ROWS / 256;                    // 128
    const dim3 G512(M_ROWS / 128, CC / 128);            // (256, 4)
    const dim3 G128(M_ROWS / 128, 1);                   // (256, 1)
    const dim3 TB(32, 8);
    const dim3 WG(CC / 32, CC / 32, L_NUM);

    // ---- weight prep: bf16 + transpose, once ----
    wtrans_kernel<<<WG, TB, 0, stream>>>(Wv,   Wvt);
    wtrans_kernel<<<WG, TB, 0, stream>>>(Wout, Woutt);
    wtrans_kernel<<<WG, TB, 0, stream>>>(W1,   W1t);
    wtrans_kernel<<<WG, TB, 0, stream>>>(W2,   W2t);
    wtrans_kernel<<<dim3(CC / 32, CC / 32, 1), TB, 0, stream>>>(peW2, peW2t);
    wcat_kernel<<<(L_NUM * 128 * CC) / 256, 256, 0, stream>>>(Woff, Waw, Wcatt);
    bcat_kernel<<<3, 256, 0, stream>>>(boff, baw, bcat);

    // ---- stage 0: voxelize / pack ----
    mask_kernel <<<BN_BLK, 256, 0, stream>>>(lidar, maskb);
    order_kernel<<<BB, 1024, 0, stream>>>(maskb, order, numV);
    pack_q_kernel<<<EW, 256, 0, stream>>>(lidar, order, numV, qbuf);
    ref_kernel  <<<BN_BLK, 256, 0, stream>>>(order, numV, refb);
    bevtobf_kernel<<<dim3(NN / 32, CC / 32, BB), TB, 0, stream>>>(bev, featbf);

    // ---- positional embedding: relu(BN(ref@peW1)) @ peW2 -> qpos ----
    pe_kernel<<<EW, 256, 0, stream>>>(refb, peW1, bng, bnb, bnm, bnv, xh);
    gemm_bf16_kernel<false, false, false, false><<<G512, 256, 0, stream>>>(
        xh, peW2t, nullptr, nullptr, qpos, nullptr, M_ROWS, CC, CC);

    // ---- 6 transformer layers ----
    for (int i = 0; i < L_NUM; ++i) {
        const ushortx* Wvi = Wvt   + (size_t)i * CC * CC;
        const ushortx* Wci = Wcatt + (size_t)i * 128 * CC;
        const ushortx* Woi = Woutt + (size_t)i * CC * CC;
        const ushortx* W1i = W1t   + (size_t)i * CC * CC;
        const ushortx* W2i = W2t   + (size_t)i * CC * CC;
        const float*   bvi = bv    + (size_t)i * CC;
        const float*   bci = bcat  + (size_t)i * 128;
        const float*   boi = bout  + (size_t)i * CC;

        // x = q + q_pos (bf16)
        add_kernel<<<EW, 256, 0, stream>>>(qbuf, qpos, xh);
        // v = feat_flat @ Wv + bv  (f32 out)
        gemm_bf16_kernel<false, true, false, false><<<G512, 256, 0, stream>>>(
            featbf, Wvi, bvi, nullptr, vbuf, nullptr, M_ROWS, CC, CC);
        // [off | aw] = x @ Wcat + bcat
        gemm_bf16_kernel<false, true, false, false><<<G128, 256, 0, stream>>>(
            xh, Wci, bci, nullptr, offaw, nullptr, M_ROWS, CC, 128);
        // deformable sampling (softmax fused) -> bf16
        sample_kernel<<<M_ROWS, 256, 0, stream>>>(vbuf, offaw, refb, samph);
        // t1 = samp @ Wout + bout + q   (vbuf reused as scratch)
        gemm_bf16_kernel<false, true, true, false><<<G512, 256, 0, stream>>>(
            samph, Woi, boi, qbuf, vbuf, nullptr, M_ROWS, CC, CC);
        layernorm_kernel<<<M_ROWS, 256, 0, stream>>>(
            vbuf, ln1g + (size_t)i * CC, ln1b + (size_t)i * CC, qbuf, qh);
        // h = relu(q @ W1)  (bf16 out, straight into next GEMM)
        gemm_bf16_kernel<true, false, false, true><<<G512, 256, 0, stream>>>(
            qh, W1i, nullptr, nullptr, nullptr, xh, M_ROWS, CC, CC);
        // t1 = h @ W2 + q
        gemm_bf16_kernel<false, false, true, false><<<G512, 256, 0, stream>>>(
            xh, W2i, nullptr, qbuf, vbuf, nullptr, M_ROWS, CC, CC);
        layernorm_kernel<<<M_ROWS, 256, 0, stream>>>(
            vbuf, ln2g + (size_t)i * CC, ln2b + (size_t)i * CC, qbuf, qh);
    }

    // ---- scatter back to dense (B,C,H,W) ----
    scatter_kernel<<<EW, 256, 0, stream>>>(qbuf, order, numV, out);
}
